// GPT_59485297050287
// MI455X (gfx1250) — matrix operand order repaired and checked by reading, not compiled, not run on hardware
//
#include <hip/hip_runtime.h>
#include <stdint.h>

// ---------------------------------------------------------------------------
// CDNA5 / gfx1250 (MI455X) implementation. wave32, WMMA bf16 16x16x32 path.
// All matmuls -> v_wmma_f32_16x16x32_bf16 (f32 accumulate). Norms / softmax /
// residuals stay f32. Workspace arena (~360 MB) holds bf16 weight repacks and
// activation scratch; attention is processed per batch to halve score memory.
// GEMM waves own 32x64 output tiles (8 accumulators) to cut VMEM issue per
// WMMA from 2.5 to 1.5 loads; global_prefetch_b8 pulls weight lines ahead.
// ---------------------------------------------------------------------------

typedef __bf16 bf16;
typedef __attribute__((ext_vector_type(16))) __bf16 v16bf;
typedef __attribute__((ext_vector_type(8)))  float  v8f;

#define CN_EPS  1e-8f
#define RMS_EPS 1.1920929e-07f

__device__ __forceinline__ bf16 f2bf(float f) {
  uint32_t u = __float_as_uint(f);
  uint32_t r = (u + 0x7FFFu + ((u >> 16) & 1u)) >> 16;   // round-to-nearest-even
  uint16_t h = (uint16_t)r;
  bf16 b;
  __builtin_memcpy(&b, &h, 2);
  return b;
}

union V16U { v16bf v; uint4 q[2]; };

// A-tile 16x32 bf16 (row-major source, ld = lda elements).
// ISA layout: lane<16 holds row m=lane, K-octets {h*8.., 16+h*8..} with h=lane>>4.
__device__ __forceinline__ v16bf load_a16x32(const bf16* __restrict__ A, int lda,
                                             int m0, int k0, int lane) {
  const int m = m0 + (lane & 15);
  const int h = (lane >> 4) & 1;
  const bf16* p = A + (size_t)m * lda + (k0 + h * 8);
  V16U u;
  u.q[0] = *(const uint4*)(p);
  u.q[1] = *(const uint4*)(p + 16);
  return u.v;
}

// B-tile 32x16 bf16 where B = Wrowsᵀ: lane n holds K-octets {h*8.., 16+h*8..} of
// source row (n0 + lane&15), h = lane>>4.
__device__ __forceinline__ v16bf load_b32x16(const bf16* __restrict__ W, int ldw,
                                             int n0, int k0, int lane) {
  const int n = n0 + (lane & 15);
  const bf16* p = W + (size_t)n * ldw + (k0 + ((lane >> 4) & 1) * 8);
  V16U u;
  u.q[0] = *(const uint4*)(p);
  u.q[1] = *(const uint4*)(p + 16);
  return u.v;
}

__device__ __forceinline__ v8f wmma_bf16(v16bf a, v16bf b, v8f c) {
  return __builtin_amdgcn_wmma_f32_16x16x32_bf16(false, a, false, b, (short)0, c,
                                                 false, false);
}

#define V8F_ZERO {0.f,0.f,0.f,0.f,0.f,0.f,0.f,0.f}

// ---------------------------------------------------------------------------
// Generic GEMM:  C[M,N] (f32/bf16) = A[M,K]_bf16 @ W[N,K]_bf16ᵀ
// MODE 0: store f32   MODE 1: += into f32   MODE 2: relu(x)^2 -> bf16
// One wave owns a 32x64 output strip (2 m-tiles x 4 n-tiles, 8 accumulators):
// per K-step 12 b128 loads feed 8 WMMAs (B reused 2x, A reused 4x).
// ---------------------------------------------------------------------------
template<int MODE>
__global__ void __launch_bounds__(256)
k_gemm(const bf16* __restrict__ A, const bf16* __restrict__ W,
       void* __restrict__ Cout, int M, int N, int K) {
  const int lane = threadIdx.x & 31;
  const int wave = threadIdx.x >> 5;
  const int tiles_n = N >> 6;
  const int tiles_m = M >> 5;
  const int tile = blockIdx.x * 8 + wave;
  if (tile >= tiles_m * tiles_n) return;
  const int mt = tile / tiles_n;
  const int nt = tile - mt * tiles_n;
  const int m0 = mt << 5;
  const int n0 = nt << 6;
  v8f acc00 = V8F_ZERO, acc01 = V8F_ZERO, acc02 = V8F_ZERO, acc03 = V8F_ZERO;
  v8f acc10 = V8F_ZERO, acc11 = V8F_ZERO, acc12 = V8F_ZERO, acc13 = V8F_ZERO;
  for (int k0 = 0; k0 < K; k0 += 32) {
    if (k0 + 64 < K) {   // pull next-next K-step lines toward L2/WGP$
      __builtin_prefetch(A + (size_t)(m0 + (lane & 15)) * K + k0 + 64, 0, 3);
      __builtin_prefetch(W + (size_t)(n0 + (lane & 15)) * K + k0 + 64, 0, 3);
    }
    v16bf a0 = load_a16x32(A, K, m0,      k0, lane);
    v16bf a1 = load_a16x32(A, K, m0 + 16, k0, lane);
    v16bf b0 = load_b32x16(W, K, n0 +  0, k0, lane);
    v16bf b1 = load_b32x16(W, K, n0 + 16, k0, lane);
    v16bf b2 = load_b32x16(W, K, n0 + 32, k0, lane);
    v16bf b3 = load_b32x16(W, K, n0 + 48, k0, lane);
    acc00 = wmma_bf16(a0, b0, acc00);
    acc01 = wmma_bf16(a0, b1, acc01);
    acc02 = wmma_bf16(a0, b2, acc02);
    acc03 = wmma_bf16(a0, b3, acc03);
    acc10 = wmma_bf16(a1, b0, acc10);
    acc11 = wmma_bf16(a1, b1, acc11);
    acc12 = wmma_bf16(a1, b2, acc12);
    acc13 = wmma_bf16(a1, b3, acc13);
  }
  const int c = lane & 15;
  v8f accs[2][4] = {{acc00, acc01, acc02, acc03}, {acc10, acc11, acc12, acc13}};
#pragma unroll
  for (int mi = 0; mi < 2; ++mi) {
    const int rb = m0 + mi * 16 + ((lane >> 4) << 3);
#pragma unroll
    for (int j = 0; j < 4; ++j) {
      const int col = n0 + j * 16 + c;
#pragma unroll
      for (int v = 0; v < 8; ++v) {
        const size_t i = (size_t)(rb + v) * N + col;
        if constexpr (MODE == 0) {
          ((float*)Cout)[i] = accs[mi][j][v];
        } else if constexpr (MODE == 1) {
          ((float*)Cout)[i] += accs[mi][j][v];
        } else {
          float x = accs[mi][j][v];
          x = x > 0.f ? x : 0.f;
          ((bf16*)Cout)[i] = f2bf(x * x);
        }
      }
    }
  }
}

// ---------------------------------------------------------------------------
// Attention score kernel. Qc=[qr|qi], Qc2=[qi|-qr], Kc=[kr|ki], K=128.
// Wave computes a 16x32 (t,s) strip (4 accumulators); fully-masked strips
// skipped (softmax applies the exact mask).
// ---------------------------------------------------------------------------
__global__ void __launch_bounds__(256)
k_scores(const bf16* __restrict__ Qc, const bf16* __restrict__ Qc2,
         const bf16* __restrict__ Kc, float* __restrict__ Sr,
         float* __restrict__ Si, int T, int win) {
  const int lane = threadIdx.x & 31;
  const int wave = threadIdx.x >> 5;
  const int s0 = (blockIdx.x * 8 + wave) << 5;    // 32-wide s strip
  const int t0 = blockIdx.y << 4;
  const int h  = blockIdx.z;
  if (s0 > t0 + 15) return;                 // fully above diagonal
  if (t0 > s0 + win + 30) return;           // fully outside window
  const bf16* q  = Qc  + (size_t)h * T * 128;
  const bf16* q2 = Qc2 + (size_t)h * T * 128;
  const bf16* kk = Kc  + (size_t)h * T * 128;
  float* sr = Sr + (size_t)h * T * T;
  float* si = Si + (size_t)h * T * T;
  v8f ar0 = V8F_ZERO, ai0 = V8F_ZERO, ar1 = V8F_ZERO, ai1 = V8F_ZERO;
#pragma unroll
  for (int k0 = 0; k0 < 128; k0 += 32) {
    v16bf a1 = load_a16x32(q,  128, t0, k0, lane);
    v16bf a2 = load_a16x32(q2, 128, t0, k0, lane);
    v16bf b0 = load_b32x16(kk, 128, s0,      k0, lane);
    v16bf b1 = load_b32x16(kk, 128, s0 + 16, k0, lane);
    ar0 = wmma_bf16(a1, b0, ar0);
    ai0 = wmma_bf16(a2, b0, ai0);
    ar1 = wmma_bf16(a1, b1, ar1);
    ai1 = wmma_bf16(a2, b1, ai1);
  }
  const int rb = t0 + ((lane >> 4) << 3);
  const int c  = lane & 15;
#pragma unroll
  for (int v = 0; v < 8; ++v) {
    const size_t row = (size_t)(rb + v) * T;
    sr[row + s0 + c]      = ar0[v];
    si[row + s0 + c]      = ai0[v];
    sr[row + s0 + 16 + c] = ar1[v];
    si[row + s0 + 16 + c] = ai1[v];
  }
}

// ---------------------------------------------------------------------------
// Row softmax of |S| (masked entries contribute exp(sqrt(CN_EPS))), writes
// hr = sr*A, hi = si*A as bf16. One 256-thread block per (h,t) row, T=1024.
// ---------------------------------------------------------------------------
__global__ void __launch_bounds__(256)
k_softmax(const float* __restrict__ Sr, const float* __restrict__ Si,
          bf16* __restrict__ Hr, bf16* __restrict__ Hi, int T, int win,
          float scale) {
  __shared__ float sm[256];
  const int t   = blockIdx.x;
  const int h   = blockIdx.y;
  const int tid = threadIdx.x;
  const size_t row = ((size_t)h * T + t) * T;
  float vr[4], vi[4], sab[4];
  float mx = 0.f;
#pragma unroll
  for (int j = 0; j < 4; ++j) {
    const int s = tid + j * 256;
    const bool keep = (s <= t) && ((t - s) < win);
    float a = keep ? Sr[row + s] * scale : 0.f;
    float b = keep ? Si[row + s] * scale : 0.f;
    vr[j] = a; vi[j] = b;
    sab[j] = sqrtf(a * a + b * b + CN_EPS);
    mx = fmaxf(mx, sab[j]);
  }
  sm[tid] = mx; __syncthreads();
  for (int s = 128; s > 0; s >>= 1) {
    if (tid < s) sm[tid] = fmaxf(sm[tid], sm[tid + s]);
    __syncthreads();
  }
  mx = sm[0]; __syncthreads();
  float se = 0.f;
#pragma unroll
  for (int j = 0; j < 4; ++j) { sab[j] = expf(sab[j] - mx); se += sab[j]; }
  sm[tid] = se; __syncthreads();
  for (int s = 128; s > 0; s >>= 1) {
    if (tid < s) sm[tid] += sm[tid + s];
    __syncthreads();
  }
  const float inv = 1.f / sm[0];
#pragma unroll
  for (int j = 0; j < 4; ++j) {
    const int s = tid + j * 256;
    const float A = sab[j] * inv;
    Hr[row + s] = f2bf(vr[j] * A);
    Hi[row + s] = f2bf(vi[j] * A);
  }
}

// ---------------------------------------------------------------------------
// Attention output: dz_r = hr@vi + hi@vr ; dz_i = hi@vi - hr@vr.
// Wave owns a 32x16 (t,d) tile -> 8 accumulators; per K-step 12 loads feed
// 8 WMMAs. K-range clipped to the causal/window band. V stored transposed
// (d-major) so B-tiles are contiguous. Writes bf16 into (b,t,h,d) layout.
// ---------------------------------------------------------------------------
__global__ void __launch_bounds__(128)
k_attnout(const bf16* __restrict__ Hr, const bf16* __restrict__ Hi,
          const bf16* __restrict__ Vit, const bf16* __restrict__ Vrt,
          bf16* __restrict__ dzr, bf16* __restrict__ dzi, int T, int win) {
  const int lane = threadIdx.x & 31;
  const int wave = threadIdx.x >> 5;     // n-tile within head dim (HD=64)
  const int t0 = blockIdx.x << 5;        // 32-row t strip
  const int h  = blockIdx.y;
  const bf16* hr  = Hr  + (size_t)h * T * T;
  const bf16* hi  = Hi  + (size_t)h * T * T;
  const bf16* bvi = Vit + (size_t)h * 64 * T;
  const bf16* bvr = Vrt + (size_t)h * 64 * T;
  const int d0 = wave << 4;
  v8f rvi0 = V8F_ZERO, ivr0 = V8F_ZERO, ivi0 = V8F_ZERO, rvr0 = V8F_ZERO;
  v8f rvi1 = V8F_ZERO, ivr1 = V8F_ZERO, ivi1 = V8F_ZERO, rvr1 = V8F_ZERO;
  int klo = t0 - win + 1; if (klo < 0) klo = 0; klo &= ~31;
  const int khi = (t0 + 31) & ~31;
  for (int k0 = klo; k0 <= khi; k0 += 32) {
    v16bf a_r0 = load_a16x32(hr, T, t0,      k0, lane);
    v16bf a_i0 = load_a16x32(hi, T, t0,      k0, lane);
    v16bf a_r1 = load_a16x32(hr, T, t0 + 16, k0, lane);
    v16bf a_i1 = load_a16x32(hi, T, t0 + 16, k0, lane);
    v16bf b_i  = load_b32x16(bvi, T, d0, k0, lane);
    v16bf b_r  = load_b32x16(bvr, T, d0, k0, lane);
    rvi0 = wmma_bf16(a_r0, b_i, rvi0);
    ivr0 = wmma_bf16(a_i0, b_r, ivr0);
    ivi0 = wmma_bf16(a_i0, b_i, ivi0);
    rvr0 = wmma_bf16(a_r0, b_r, rvr0);
    rvi1 = wmma_bf16(a_r1, b_i, rvi1);
    ivr1 = wmma_bf16(a_i1, b_r, ivr1);
    ivi1 = wmma_bf16(a_i1, b_i, ivi1);
    rvr1 = wmma_bf16(a_r1, b_r, rvr1);
  }
  const int rb = (lane >> 4) << 3;
  const int c  = lane & 15;
#pragma unroll
  for (int v = 0; v < 8; ++v) {
    const int ta = t0 + rb + v;
    const int tb = ta + 16;
    const size_t oa = (size_t)ta * 768 + (size_t)h * 64 + d0 + c;
    const size_t ob = (size_t)tb * 768 + (size_t)h * 64 + d0 + c;
    dzr[oa] = f2bf(rvi0[v] + ivr0[v]);
    dzi[oa] = f2bf(ivi0[v] - rvr0[v]);
    dzr[ob] = f2bf(rvi1[v] + ivr1[v]);
    dzi[ob] = f2bf(ivi1[v] - rvr1[v]);
  }
}

// ---------------------------------------------------------------------------
// RoPE (first 32 dims) + per-row complex_norm (scalar scale) + bf16 packing of
// Qc=[qr|qi], Qc2=[qi|-qr], Kc=[kr|ki]. One wave per (h,t) row; lane owns
// dims d=lane (roped) and d=lane+32.
// ---------------------------------------------------------------------------
__global__ void __launch_bounds__(256)
k_rope_norm_pack(const float* __restrict__ qr, const float* __restrict__ qi,
                 const float* __restrict__ kr, const float* __restrict__ ki,
                 bf16* __restrict__ Qc, bf16* __restrict__ Qc2,
                 bf16* __restrict__ Kc, int T) {
  const int lane = threadIdx.x & 31;
  const int wave = threadIdx.x >> 5;
  const int t = blockIdx.x * 8 + wave;
  const int h = blockIdx.y;
  const size_t base = (size_t)t * 768 + (size_t)h * 64;
  // inv_freq = 10000^(-lane/32) = exp(-lane * ln(10000)/32)
  const float ang = (float)t * expf(-(float)lane * 0.28782313662425574f);
  float sn, cs;
  sincosf(ang, &sn, &cs);

  // ---- Q ----
  float q0r = qr[base + lane],      q0i = qi[base + lane];
  float q1r = qr[base + lane + 32], q1i = qi[base + lane + 32];
  { float tr = q0r * cs - q0i * sn, ti = q0r * sn + q0i * cs; q0r = tr; q0i = ti; }
  float sumq = q0r*q0r + q0i*q0i + q1r*q1r + q1i*q1i;
#pragma unroll
  for (int m = 16; m > 0; m >>= 1) sumq += __shfl_xor(sumq, m, 32);
  const float sq = rsqrtf(sumq * (1.f / 64.f) + CN_EPS + RMS_EPS);
  bf16* qrow  = Qc  + ((size_t)h * T + t) * 128;
  bf16* q2row = Qc2 + ((size_t)h * T + t) * 128;
  qrow[lane]       = f2bf(q0r * sq);
  qrow[lane + 32]  = f2bf(q1r * sq);
  qrow[lane + 64]  = f2bf(q0i * sq);
  qrow[lane + 96]  = f2bf(q1i * sq);
  q2row[lane]      = f2bf(q0i * sq);
  q2row[lane + 32] = f2bf(q1i * sq);
  q2row[lane + 64] = f2bf(-q0r * sq);
  q2row[lane + 96] = f2bf(-q1r * sq);

  // ---- K ----
  float k0r = kr[base + lane],      k0i = ki[base + lane];
  float k1r = kr[base + lane + 32], k1i = ki[base + lane + 32];
  { float tr = k0r * cs - k0i * sn, ti = k0r * sn + k0i * cs; k0r = tr; k0i = ti; }
  float sumk = k0r*k0r + k0i*k0i + k1r*k1r + k1i*k1i;
#pragma unroll
  for (int m = 16; m > 0; m >>= 1) sumk += __shfl_xor(sumk, m, 32);
  const float sk = rsqrtf(sumk * (1.f / 64.f) + CN_EPS + RMS_EPS);
  bf16* krow = Kc + ((size_t)h * T + t) * 128;
  krow[lane]      = f2bf(k0r * sk);
  krow[lane + 32] = f2bf(k1r * sk);
  krow[lane + 64] = f2bf(k0i * sk);
  krow[lane + 96] = f2bf(k1i * sk);
}

// Pack V (b fixed) from (t,h,d) f32 into transposed (h,d,t) bf16.
__global__ void __launch_bounds__(256)
k_pack_vt(const float* __restrict__ vr, const float* __restrict__ vi,
          bf16* __restrict__ Vrt, bf16* __restrict__ Vit, int T) {
  const int i = blockIdx.x * 256 + threadIdx.x;          // over NH*T*HD
  if (i >= 12 * 1024 * 64) return;
  const int h = i / (T * 64);
  const int rem = i - h * T * 64;
  const int t = rem >> 6;
  const int d = rem & 63;
  const size_t src = (size_t)t * 768 + (size_t)h * 64 + d;
  const size_t dst = ((size_t)h * 64 + d) * T + t;
  Vrt[dst] = f2bf(vr[src]);
  Vit[dst] = f2bf(vi[src]);
}

// ---------------------------------------------------------------------------
// Token-wise complex_norm variants (block of 256 per token, D=768).
// ---------------------------------------------------------------------------
__device__ __forceinline__ float block_sum(float v, float* sm) {
  const int tid = threadIdx.x;
  sm[tid] = v; __syncthreads();
  for (int s = 128; s > 0; s >>= 1) {
    if (tid < s) sm[tid] += sm[tid + s];
    __syncthreads();
  }
  float r = sm[0]; __syncthreads();
  return r;
}

__global__ void __launch_bounds__(256)
k_cnorm_pack(const float* __restrict__ xr, const float* __restrict__ xi,
             bf16* __restrict__ Xcat) {
  __shared__ float sm[256];
  const int tok = blockIdx.x, tid = threadIdx.x;
  const float* pr = xr + (size_t)tok * 768;
  const float* pi = xi + (size_t)tok * 768;
  float vr[3], vi[3], part = 0.f;
#pragma unroll
  for (int j = 0; j < 3; ++j) {
    const int d = tid + j * 256;
    vr[j] = pr[d]; vi[j] = pi[d];
    part += vr[j] * vr[j] + vi[j] * vi[j];
  }
  const float s = rsqrtf(block_sum(part, sm) * (1.f / 768.f) + CN_EPS + RMS_EPS);
  bf16* row = Xcat + (size_t)tok * 1536;
#pragma unroll
  for (int j = 0; j < 3; ++j) {
    const int d = tid + j * 256;
    row[d]       = f2bf(vr[j] * s);
    row[768 + d] = f2bf(vi[j] * s);
  }
}

__global__ void __launch_bounds__(256)
k_cnorm_mlp_pack(const float* __restrict__ xr, const float* __restrict__ xi,
                 float* __restrict__ rmag, bf16* __restrict__ rbf,
                 float* __restrict__ s_tok) {
  __shared__ float sm[256];
  const int tok = blockIdx.x, tid = threadIdx.x;
  const float* pr = xr + (size_t)tok * 768;
  const float* pi = xi + (size_t)tok * 768;
  float vr[3], vi[3], part = 0.f;
#pragma unroll
  for (int j = 0; j < 3; ++j) {
    const int d = tid + j * 256;
    vr[j] = pr[d]; vi[j] = pi[d];
    part += vr[j] * vr[j] + vi[j] * vi[j];
  }
  const float s = rsqrtf(block_sum(part, sm) * (1.f / 768.f) + CN_EPS + RMS_EPS);
  if (tid == 0) s_tok[tok] = s;
#pragma unroll
  for (int j = 0; j < 3; ++j) {
    const int d = tid + j * 256;
    const float a = vr[j] * s, b = vi[j] * s;
    const float r = sqrtf(a * a + b * b + CN_EPS);
    rmag[(size_t)tok * 768 + d] = r;
    rbf [(size_t)tok * 768 + d] = f2bf(r);
  }
}

__global__ void __launch_bounds__(256)
k_cnorm_final(const float* __restrict__ xr, const float* __restrict__ xi,
              bf16* __restrict__ xfin) {
  __shared__ float sm[256];
  const int tok = blockIdx.x, tid = threadIdx.x;
  const float* pr = xr + (size_t)tok * 768;
  const float* pi = xi + (size_t)tok * 768;
  float vr[3], part = 0.f;
#pragma unroll
  for (int j = 0; j < 3; ++j) {
    const int d = tid + j * 256;
    vr[j] = pr[d];
    const float b = pi[d];
    part += vr[j] * vr[j] + b * b;
  }
  const float s = rsqrtf(block_sum(part, sm) * (1.f / 768.f) + CN_EPS + RMS_EPS);
#pragma unroll
  for (int j = 0; j < 3; ++j) {
    const int d = tid + j * 256;
    xfin[(size_t)tok * 768 + d] = f2bf(vr[j] * s);
  }
}

// MLP epilogue: z=r+delta; r_new = z>20 ? z : softplus(z); residual update.
__global__ void __launch_bounds__(256)
k_mlp_final(float* __restrict__ xr, float* __restrict__ xi,
            const float* __restrict__ rmag, const float* __restrict__ delta,
            const float* __restrict__ s_tok, int n) {
  const int i = blockIdx.x * 256 + threadIdx.x;
  if (i >= n) return;
  const float s = s_tok[i / 768];
  const float r = rmag[i];
  const float z = r + delta[i];
  const float rn = (z > 20.f) ? z : log1pf(expf(z));
  const float f = s * (rn - r) / r;
  xr[i] += xr[i] * f;
  xi[i] += xi[i] * f;
}

// Embedding gather.
__global__ void __launch_bounds__(256)
k_embed(const int* __restrict__ idx, const float* __restrict__ wte,
        float* __restrict__ xr, float* __restrict__ xi, int n) {
  const int i = blockIdx.x * 256 + threadIdx.x;
  if (i >= n) return;
  const int tok = i / 768, d = i - tok * 768;
  xr[i] = wte[(size_t)idx[tok] * 768 + d];
  xi[i] = 0.f;
}

// Pack complex-linear weights: w (1536x768) -> Wr2=[wr|-wi], Wi2=[wi|wr] (768x1536).
__global__ void __launch_bounds__(256)
k_pack_cw(const float* __restrict__ w, bf16* __restrict__ wr2,
          bf16* __restrict__ wi2) {
  const int i = blockIdx.x * 256 + threadIdx.x;
  if (i >= 768 * 1536) return;
  const int n = i / 1536, k = i - n * 1536;
  if (k < 768) {
    wr2[i] = f2bf(w[(size_t)n * 768 + k]);
    wi2[i] = f2bf(w[(size_t)(768 + n) * 768 + k]);
  } else {
    const int kk = k - 768;
    wr2[i] = f2bf(-w[(size_t)(768 + n) * 768 + kk]);
    wi2[i] = f2bf( w[(size_t)n * 768 + kk]);
  }
}

__global__ void __launch_bounds__(256)
k_cast_bf(const float* __restrict__ s, bf16* __restrict__ d, size_t n) {
  const size_t i = (size_t)blockIdx.x * 256 + threadIdx.x;
  if (i < n) d[i] = f2bf(s[i]);
}

// ---------------------------------------------------------------------------
// Host orchestration
// ---------------------------------------------------------------------------
static void launch_gemm(int mode, const bf16* A, const bf16* W, void* C,
                        int M, int N, int K, hipStream_t s) {
  const int tiles = (M / 32) * (N / 64);
  const int blocks = (tiles + 7) / 8;
  if (mode == 0)      k_gemm<0><<<blocks, 256, 0, s>>>(A, W, C, M, N, K);
  else if (mode == 1) k_gemm<1><<<blocks, 256, 0, s>>>(A, W, C, M, N, K);
  else                k_gemm<2><<<blocks, 256, 0, s>>>(A, W, C, M, N, K);
}

extern "C" void kernel_launch(void* const* d_in, const int* in_sizes, int n_in,
                              void* d_out, int out_size, void* d_ws, size_t ws_size,
                              hipStream_t stream) {
  (void)in_sizes; (void)n_in; (void)out_size; (void)ws_size;
  constexpr int Tc = 1024, NLc = 2, NHc = 12, Dc = 768, Vc = 32000;
  constexpr int Mtok = 2 * Tc;                 // B*T = 2048
  const int wins[NLc] = {512, 1024};

  const int*   idx   = (const int*)d_in[0];
  const float* wte   = (const float*)d_in[1];
  const float* Wq    = (const float*)d_in[2];
  const float* Wk    = (const float*)d_in[3];
  const float* Wv    = (const float*)d_in[4];
  const float* Wo    = (const float*)d_in[5];
  const float* Wfc   = (const float*)d_in[6];
  const float* Wproj = (const float*)d_in[7];
  const float* Wlm   = (const float*)d_in[8];

  char* base = (char*)d_ws;
  size_t off = 0;
  auto alloc = [&](size_t bytes) -> void* {
    void* p = base + off;
    off = (off + bytes + 255) & ~(size_t)255;
    return p;
  };

  // bf16 weight repacks
  const size_t n_cw = (size_t)NLc * Dc * 2 * Dc;        // 2,359,296 per tensor
  bf16* wq_r2 = (bf16*)alloc(n_cw * 2);
  bf16* wq_i2 = (bf16*)alloc(n_cw * 2);
  bf16* wk_r2 = (bf16*)alloc(n_cw * 2);
  bf16* wk_i2 = (bf16*)alloc(n_cw * 2);
  bf16* wv_r2 = (bf16*)alloc(n_cw * 2);
  bf16* wv_i2 = (bf16*)alloc(n_cw * 2);
  bf16* wo_bf    = (bf16*)alloc((size_t)NLc * Dc * Dc * 2);
  bf16* wfc_bf   = (bf16*)alloc((size_t)NLc * 4 * Dc * Dc * 2);
  bf16* wproj_bf = (bf16*)alloc((size_t)NLc * Dc * 4 * Dc * 2);
  bf16* lm_bf    = (bf16*)alloc((size_t)Vc * Dc * 2);

  // activations
  float* xr = (float*)alloc((size_t)Mtok * Dc * 4);
  float* xi = (float*)alloc((size_t)Mtok * Dc * 4);
  bf16*  Xcat = (bf16*)alloc((size_t)Mtok * 2 * Dc * 2);
  float* qr = (float*)alloc((size_t)Mtok * Dc * 4);
  float* qi = (float*)alloc((size_t)Mtok * Dc * 4);
  float* kr = (float*)alloc((size_t)Mtok * Dc * 4);
  float* ki = (float*)alloc((size_t)Mtok * Dc * 4);
  float* vr = (float*)alloc((size_t)Mtok * Dc * 4);
  float* vi = (float*)alloc((size_t)Mtok * Dc * 4);
  bf16* Qc  = (bf16*)alloc((size_t)NHc * Tc * 128 * 2);
  bf16* Qc2 = (bf16*)alloc((size_t)NHc * Tc * 128 * 2);
  bf16* Kc  = (bf16*)alloc((size_t)NHc * Tc * 128 * 2);
  bf16* Vrt = (bf16*)alloc((size_t)NHc * 64 * Tc * 2);
  bf16* Vit = (bf16*)alloc((size_t)NHc * 64 * Tc * 2);
  float* Sr = (float*)alloc((size_t)NHc * Tc * Tc * 4);   // per-batch reuse
  float* Si = (float*)alloc((size_t)NHc * Tc * Tc * 4);
  bf16* Hr  = (bf16*)alloc((size_t)NHc * Tc * Tc * 2);
  bf16* Hi  = (bf16*)alloc((size_t)NHc * Tc * Tc * 2);
  bf16* dzr = (bf16*)alloc((size_t)Mtok * Dc * 2);
  bf16* dzi = (bf16*)alloc((size_t)Mtok * Dc * 2);
  float* rmag  = (float*)alloc((size_t)Mtok * Dc * 4);
  bf16*  rbf   = (bf16*)alloc((size_t)Mtok * Dc * 2);
  float* s_tok = (float*)alloc((size_t)Mtok * 4);
  bf16*  hbf   = (bf16*)alloc((size_t)Mtok * 4 * Dc * 2);
  float* delta = (float*)alloc((size_t)Mtok * Dc * 4);
  bf16*  xfin  = (bf16*)alloc((size_t)Mtok * Dc * 2);

  // ---- weight packing (deterministic, rebuilt every call) ----
  const int cw_blocks = (768 * 1536 + 255) / 256;
  for (int l = 0; l < NLc; ++l) {
    const size_t ws = (size_t)l * 2 * Dc * Dc;
    const size_t ds = (size_t)l * Dc * 2 * Dc;
    k_pack_cw<<<cw_blocks, 256, 0, stream>>>(Wq + ws, wq_r2 + ds, wq_i2 + ds);
    k_pack_cw<<<cw_blocks, 256, 0, stream>>>(Wk + ws, wk_r2 + ds, wk_i2 + ds);
    k_pack_cw<<<cw_blocks, 256, 0, stream>>>(Wv + ws, wv_r2 + ds, wv_i2 + ds);
  }
  {
    size_t n;
    n = (size_t)NLc * Dc * Dc;
    k_cast_bf<<<(unsigned)((n + 255) / 256), 256, 0, stream>>>(Wo, wo_bf, n);
    n = (size_t)NLc * 4 * Dc * Dc;
    k_cast_bf<<<(unsigned)((n + 255) / 256), 256, 0, stream>>>(Wfc, wfc_bf, n);
    k_cast_bf<<<(unsigned)((n + 255) / 256), 256, 0, stream>>>(Wproj, wproj_bf, n);
    n = (size_t)Vc * Dc;
    k_cast_bf<<<(unsigned)((n + 255) / 256), 256, 0, stream>>>(Wlm, lm_bf, n);
  }

  // ---- embedding ----
  k_embed<<<(Mtok * Dc + 255) / 256, 256, 0, stream>>>(idx, wte, xr, xi, Mtok * Dc);

  // ---- layers ----
  for (int l = 0; l < NLc; ++l) {
    const int win = wins[l];
    const size_t ds = (size_t)l * Dc * 2 * Dc;

    k_cnorm_pack<<<Mtok, 256, 0, stream>>>(xr, xi, Xcat);
    launch_gemm(0, Xcat, wq_r2 + ds, qr, Mtok, Dc, 2 * Dc, stream);
    launch_gemm(0, Xcat, wq_i2 + ds, qi, Mtok, Dc, 2 * Dc, stream);
    launch_gemm(0, Xcat, wk_r2 + ds, kr, Mtok, Dc, 2 * Dc, stream);
    launch_gemm(0, Xcat, wk_i2 + ds, ki, Mtok, Dc, 2 * Dc, stream);
    launch_gemm(0, Xcat, wv_r2 + ds, vr, Mtok, Dc, 2 * Dc, stream);
    launch_gemm(0, Xcat, wv_i2 + ds, vi, Mtok, Dc, 2 * Dc, stream);

    for (int b = 0; b < 2; ++b) {
      const size_t bo = (size_t)b * Tc * Dc;
      k_rope_norm_pack<<<dim3(Tc / 8, NHc), 256, 0, stream>>>(
          qr + bo, qi + bo, kr + bo, ki + bo, Qc, Qc2, Kc, Tc);
      k_pack_vt<<<(NHc * Tc * 64 + 255) / 256, 256, 0, stream>>>(
          vr + bo, vi + bo, Vrt, Vit, Tc);
      k_scores<<<dim3(Tc / 32 / 8, Tc / 16, NHc), 256, 0, stream>>>(
          Qc, Qc2, Kc, Sr, Si, Tc, win);
      k_softmax<<<dim3(Tc, NHc), 256, 0, stream>>>(Sr, Si, Hr, Hi, Tc, win, 0.125f);
      k_attnout<<<dim3(Tc / 32, NHc), 128, 0, stream>>>(
          Hr, Hi, Vit, Vrt, dzr + bo, dzi + bo, Tc, win);
    }

    const bf16* wo_l = wo_bf + (size_t)l * Dc * Dc;
    launch_gemm(1, dzr, wo_l, xr, Mtok, Dc, Dc, stream);     // xr += dz_r @ Woᵀ
    launch_gemm(1, dzi, wo_l, xi, Mtok, Dc, Dc, stream);     // xi += dz_i @ Woᵀ

    k_cnorm_mlp_pack<<<Mtok, 256, 0, stream>>>(xr, xi, rmag, rbf, s_tok);
    launch_gemm(2, rbf, wfc_bf + (size_t)l * 4 * Dc * Dc, hbf,
                Mtok, 4 * Dc, Dc, stream);                   // relu(.)^2 -> bf16
    launch_gemm(0, hbf, wproj_bf + (size_t)l * Dc * 4 * Dc, delta,
                Mtok, Dc, 4 * Dc, stream);
    k_mlp_final<<<(Mtok * Dc + 255) / 256, 256, 0, stream>>>(
        xr, xi, rmag, delta, s_tok, Mtok * Dc);
  }

  // ---- final norm + lm head ----
  k_cnorm_final<<<Mtok, 256, 0, stream>>>(xr, xi, xfin);
  launch_gemm(0, xfin, lm_bf, (float*)d_out, Mtok, Vc, Dc, stream);
}
